// MeshGraphNet_71906342470111
// MI455X (gfx1250) — compile-verified, matrix operand order checked by reading, not run on hardware
//
#include <hip/hip_runtime.h>

typedef __attribute__((ext_vector_type(16))) __bf16 v16bf;
typedef __attribute__((ext_vector_type(8)))  float  v8f;
typedef unsigned int u32;
typedef unsigned short u16;

// ---------------- helpers ----------------

// hardware fp32 -> bf16 conversion (single cvt instruction)
__device__ __forceinline__ u16 f2bf(float f) {
    union { __bf16 b; u16 s; } c;
    c.b = (__bf16)f;
    return c.s;
}

// convert float4 -> 4 bf16 and store as one 8-byte LDS write
__device__ __forceinline__ void store_bf4(u16* dst, float4 f) {
    union { __bf16 b[4]; uint2 u; } p;
    p.b[0] = (__bf16)f.x; p.b[1] = (__bf16)f.y;
    p.b[2] = (__bf16)f.z; p.b[3] = (__bf16)f.w;
    *(uint2*)dst = p.u;
}

// A fragment: 16x32 bf16 tile, lane L holds row M=L&15, half=(L>>4) selects K groups.
// VGPR v: K = kt*32 + (v>>2)*16 + half*8 + (v&3)*2, elements (K, K+1).
template<int KT>
__device__ __forceinline__ void load_afrags(const u16* lds, int ldw, int lane, v16bf* a) {
    int m = lane & 15, half = lane >> 4;
    #pragma unroll
    for (int kt = 0; kt < KT; ++kt) {
        union { u32 d[8]; v16bf v; } u;
        #pragma unroll
        for (int v = 0; v < 8; ++v) {
            int k = kt * 32 + ((v >> 2) * 16) + half * 8 + (v & 3) * 2;
            u.d[v] = *(const u32*)(lds + m * ldw + k);   // 2 consecutive bf16 = b32
        }
        a[kt] = u.v;
    }
}

// B fragment: weights pre-packed so a lane reads 8 contiguous dwords (2x b128).
__device__ __forceinline__ v16bf load_bfrag(const u32* Wp, int kt, int nt, int NT, int lane) {
    union { uint4 q[2]; v16bf v; } u;
    const uint4* p = (const uint4*)(Wp + ((((size_t)kt * NT + nt) * 32 + lane) << 3));
    u.q[0] = p[0];
    u.q[1] = p[1];
    return u.v;
}

// 16x128 GEMM (+bias, +relu) -> bf16 LDS tile [16][128]
template<int KT>
__device__ __forceinline__ void gemm_relu_lds(const u16* ldsA, int ldwA,
                                              const u32* __restrict__ Wp,
                                              const float* __restrict__ bias,
                                              u16* ldsO, int lane) {
    v16bf a[KT];
    load_afrags<KT>(ldsA, ldwA, lane, a);
    int nl = lane & 15, half = lane >> 4;
    for (int nt = 0; nt < 8; ++nt) {
        float bv = bias[nt * 16 + nl];
        v8f acc = {bv, bv, bv, bv, bv, bv, bv, bv};
        #pragma unroll
        for (int kt = 0; kt < KT; ++kt) {
            v16bf b = load_bfrag(Wp, kt, nt, 8, lane);
            acc = __builtin_amdgcn_wmma_f32_16x16x32_bf16(false, a[kt], false, b,
                                                          (short)0, acc, false, false);
        }
        #pragma unroll
        for (int r = 0; r < 8; ++r)
            ldsO[(half * 8 + r) * 128 + nt * 16 + nl] = f2bf(fmaxf(acc[r], 0.f));
    }
}

// 16x128 GEMM (+bias), fp32 accumulators kept in registers (acc[8] = 8 N-tiles)
template<int KT>
__device__ __forceinline__ void gemm_final(const u16* ldsA, int ldwA,
                                           const u32* __restrict__ Wp,
                                           const float* __restrict__ bias,
                                           int lane, v8f* acc) {
    v16bf a[KT];
    load_afrags<KT>(ldsA, ldwA, lane, a);
    int nl = lane & 15;
    for (int nt = 0; nt < 8; ++nt) {
        float bv = bias[nt * 16 + nl];
        v8f c = {bv, bv, bv, bv, bv, bv, bv, bv};
        #pragma unroll
        for (int kt = 0; kt < KT; ++kt) {
            v16bf b = load_bfrag(Wp, kt, nt, 8, lane);
            c = __builtin_amdgcn_wmma_f32_16x16x32_bf16(false, a[kt], false, b,
                                                        (short)0, c, false, false);
        }
        acc[nt] = c;
    }
}

// Row-wise LayerNorm fully in registers. Row M = (lane>>4)*8 + r; its 128 values
// live at acc[nt][r] across the 16 lanes of this lane-half -> shfl_xor reduce.
__device__ __forceinline__ void ln_inplace(v8f* acc, const float* __restrict__ g,
                                           const float* __restrict__ beta, int lane) {
    int nl = lane & 15;
    float gg[8], bb[8];
    #pragma unroll
    for (int nt = 0; nt < 8; ++nt) { gg[nt] = g[nt * 16 + nl]; bb[nt] = beta[nt * 16 + nl]; }
    #pragma unroll
    for (int r = 0; r < 8; ++r) {
        float s = 0.f, s2 = 0.f;
        #pragma unroll
        for (int nt = 0; nt < 8; ++nt) { float x = acc[nt][r]; s += x; s2 += x * x; }
        #pragma unroll
        for (int m = 1; m < 16; m <<= 1) {
            s  += __shfl_xor(s,  m, 32);
            s2 += __shfl_xor(s2, m, 32);
        }
        float mu  = s  * (1.f / 128.f);
        float var = s2 * (1.f / 128.f) - mu * mu;
        float rs  = rsqrtf(var + 1e-5f);
        #pragma unroll
        for (int nt = 0; nt < 8; ++nt)
            acc[nt][r] = (acc[nt][r] - mu) * rs * gg[nt] + bb[nt];
    }
}

// ---------------- kernels ----------------

// Pack fp32 [layers][Ksrc][Nsrc] weights into bf16 fragment order
// [l][kt][nt][lane][8 dwords], zero-padded to Kpad/Npad.
__global__ void pack_weights_kernel(const float* __restrict__ src, u32* __restrict__ dst,
                                    int layers, int Ksrc, int Kpad, int Nsrc, int Npad) {
    int NT = Npad >> 4, KT = Kpad >> 5;
    long total = (long)layers * KT * NT * 256;
    for (long i = (long)blockIdx.x * blockDim.x + threadIdx.x; i < total;
         i += (long)gridDim.x * blockDim.x) {
        int v = (int)(i & 7);
        long t = i >> 3;
        int lane = (int)(t & 31); t >>= 5;
        int nt = (int)(t % NT); t /= NT;
        int kt = (int)(t % KT);
        int l  = (int)(t / KT);
        int half = lane >> 4, nl = lane & 15;
        int K   = kt * 32 + (v >> 2) * 16 + half * 8 + (v & 3) * 2;
        int col = nt * 16 + nl;
        float f0 = (K     < Ksrc && col < Nsrc) ? src[((long)l * Ksrc + K)     * Nsrc + col] : 0.f;
        float f1 = (K + 1 < Ksrc && col < Nsrc) ? src[((long)l * Ksrc + K + 1) * Nsrc + col] : 0.f;
        dst[i] = (u32)f2bf(f0) | ((u32)f2bf(f1) << 16);
    }
}

__global__ void zero_kernel(float* __restrict__ p, long n) {
    for (long i = (long)blockIdx.x * blockDim.x + threadIdx.x; i < n;
         i += (long)gridDim.x * blockDim.x) p[i] = 0.f;
}

// Encoder: feat [R][F] -> out [R][128] = LN(MLP(F->128->128->128))
__global__ __launch_bounds__(64) void encoder_kernel(
    const float* __restrict__ feat, int F,
    const u32* __restrict__ W1p, const float* __restrict__ b1,
    const u32* __restrict__ W2p, const float* __restrict__ b2,
    const u32* __restrict__ W3p, const float* __restrict__ b3,
    const float* __restrict__ g, const float* __restrict__ beta,
    float* __restrict__ out, int R) {
    __shared__ __align__(16) u16 ldsA[2][16 * 32];
    __shared__ __align__(16) u16 ldsH1[2][16 * 128];
    __shared__ __align__(16) u16 ldsH2[2][16 * 128];
    int wave = threadIdx.x >> 5, lane = threadIdx.x & 31;
    int r0 = (blockIdx.x * 2 + wave) * 16;
    u16* A = ldsA[wave]; u16* H1 = ldsH1[wave]; u16* H2 = ldsH2[wave];
    for (int row = 0; row < 16; ++row) {
        int rr = r0 + row;
        bool ok = rr < R && lane < F;
        A[row * 32 + lane] = f2bf(ok ? feat[(long)rr * F + lane] : 0.f);
    }
    __syncthreads();
    gemm_relu_lds<1>(A, 32, W1p, b1, H1, lane);
    __syncthreads();
    gemm_relu_lds<4>(H1, 128, W2p, b2, H2, lane);
    __syncthreads();
    v8f acc[8];
    gemm_final<4>(H2, 128, W3p, b3, lane, acc);
    ln_inplace(acc, g, beta, lane);
    int nl = lane & 15, half = lane >> 4;
    for (int r = 0; r < 8; ++r) {
        int rr = r0 + half * 8 + r;
        if (rr < R)
            for (int nt = 0; nt < 8; ++nt)
                out[(long)rr * 128 + nt * 16 + nl] = acc[nt][r];
    }
}

// Edge layer: upd = LN(MLP([he | hn[s] | hn[r]])); he += upd; agg[r] += upd (atomic)
__global__ __launch_bounds__(64) void edge_layer_kernel(
    const float* __restrict__ hn, float* __restrict__ he, float* __restrict__ agg,
    const int* __restrict__ sender, const int* __restrict__ receiver,
    const u32* __restrict__ W1p, const float* __restrict__ b1,
    const u32* __restrict__ W2p, const float* __restrict__ b2,
    const u32* __restrict__ W3p, const float* __restrict__ b3,
    const float* __restrict__ g, const float* __restrict__ beta, int E) {
    __shared__ __align__(16) u16 ldsA[2][16 * 384];
    __shared__ __align__(16) u16 ldsH1[2][16 * 128];
    __shared__ __align__(16) u16 ldsH2[2][16 * 128];
    int wave = threadIdx.x >> 5, lane = threadIdx.x & 31;
    int e0 = (blockIdx.x * 2 + wave) * 16;
    u16* A = ldsA[wave]; u16* H1 = ldsH1[wave]; u16* H2 = ldsH2[wave];
    const float4 z4 = {0.f, 0.f, 0.f, 0.f};
    for (int row = 0; row < 16; ++row) {
        int e = e0 + row;
        bool ok = e < E;
        int s  = ok ? sender[e]   : 0;
        int rv = ok ? receiver[e] : 0;
        float4 a0 = ok ? ((const float4*)(he + (long)e  * 128))[lane] : z4;
        float4 a1 = ok ? ((const float4*)(hn + (long)s  * 128))[lane] : z4;
        float4 a2 = ok ? ((const float4*)(hn + (long)rv * 128))[lane] : z4;
        store_bf4(&A[row * 384 +       lane * 4], a0);
        store_bf4(&A[row * 384 + 128 + lane * 4], a1);
        store_bf4(&A[row * 384 + 256 + lane * 4], a2);
    }
    __syncthreads();
    gemm_relu_lds<12>(A, 384, W1p, b1, H1, lane);
    __syncthreads();
    gemm_relu_lds<4>(H1, 128, W2p, b2, H2, lane);
    __syncthreads();
    v8f acc[8];
    gemm_final<4>(H2, 128, W3p, b3, lane, acc);
    ln_inplace(acc, g, beta, lane);
    int nl = lane & 15, half = lane >> 4;
    for (int r = 0; r < 8; ++r) {
        int e = e0 + half * 8 + r;
        if (e < E) {
            int rv = receiver[e];
            for (int nt = 0; nt < 8; ++nt) {
                int col = nt * 16 + nl;
                float u = acc[nt][r];
                he[(long)e * 128 + col] += u;
                atomicAdd(&agg[(long)rv * 128 + col], u);
            }
        }
    }
}

// Node layer: hn += LN(MLP([hn | agg]))
__global__ __launch_bounds__(64) void node_layer_kernel(
    float* __restrict__ hn, const float* __restrict__ agg,
    const u32* __restrict__ W1p, const float* __restrict__ b1,
    const u32* __restrict__ W2p, const float* __restrict__ b2,
    const u32* __restrict__ W3p, const float* __restrict__ b3,
    const float* __restrict__ g, const float* __restrict__ beta, int N) {
    __shared__ __align__(16) u16 ldsA[2][16 * 256];
    __shared__ __align__(16) u16 ldsH1[2][16 * 128];
    __shared__ __align__(16) u16 ldsH2[2][16 * 128];
    int wave = threadIdx.x >> 5, lane = threadIdx.x & 31;
    int n0 = (blockIdx.x * 2 + wave) * 16;
    u16* A = ldsA[wave]; u16* H1 = ldsH1[wave]; u16* H2 = ldsH2[wave];
    const float4 z4 = {0.f, 0.f, 0.f, 0.f};
    for (int row = 0; row < 16; ++row) {
        int n = n0 + row;
        bool ok = n < N;
        float4 a0 = ok ? ((const float4*)(hn  + (long)n * 128))[lane] : z4;
        float4 a1 = ok ? ((const float4*)(agg + (long)n * 128))[lane] : z4;
        store_bf4(&A[row * 256 +       lane * 4], a0);
        store_bf4(&A[row * 256 + 128 + lane * 4], a1);
    }
    __syncthreads();
    gemm_relu_lds<8>(A, 256, W1p, b1, H1, lane);
    __syncthreads();
    gemm_relu_lds<4>(H1, 128, W2p, b2, H2, lane);
    __syncthreads();
    v8f acc[8];
    gemm_final<4>(H2, 128, W3p, b3, lane, acc);
    ln_inplace(acc, g, beta, lane);
    int nl = lane & 15, half = lane >> 4;
    for (int r = 0; r < 8; ++r) {
        int n = n0 + half * 8 + r;
        if (n < N)
            for (int nt = 0; nt < 8; ++nt)
                hn[(long)n * 128 + nt * 16 + nl] += acc[nt][r];
    }
}

// Decoder: out[N][3] = MLP(hn: 128->128->128->3), no LN
__global__ __launch_bounds__(64) void decoder_kernel(
    const float* __restrict__ hn,
    const u32* __restrict__ W1p, const float* __restrict__ b1,
    const u32* __restrict__ W2p, const float* __restrict__ b2,
    const u32* __restrict__ W3p, const float* __restrict__ b3,
    float* __restrict__ out, int N) {
    __shared__ __align__(16) u16 ldsA[2][16 * 128];
    __shared__ __align__(16) u16 ldsH1[2][16 * 128];
    __shared__ __align__(16) u16 ldsH2[2][16 * 128];
    int wave = threadIdx.x >> 5, lane = threadIdx.x & 31;
    int n0 = (blockIdx.x * 2 + wave) * 16;
    u16* A = ldsA[wave]; u16* H1 = ldsH1[wave]; u16* H2 = ldsH2[wave];
    const float4 z4 = {0.f, 0.f, 0.f, 0.f};
    for (int row = 0; row < 16; ++row) {
        int n = n0 + row;
        bool ok = n < N;
        float4 a0 = ok ? ((const float4*)(hn + (long)n * 128))[lane] : z4;
        store_bf4(&A[row * 128 + lane * 4], a0);
    }
    __syncthreads();
    gemm_relu_lds<4>(A, 128, W1p, b1, H1, lane);
    __syncthreads();
    gemm_relu_lds<4>(H1, 128, W2p, b2, H2, lane);
    __syncthreads();
    // final 128 -> 16 (3 real columns), single N-tile
    v16bf a[4];
    load_afrags<4>(H2, 128, lane, a);
    int nl = lane & 15, half = lane >> 4;
    float bv = (nl < 3) ? b3[nl] : 0.f;
    v8f c = {bv, bv, bv, bv, bv, bv, bv, bv};
    #pragma unroll
    for (int kt = 0; kt < 4; ++kt) {
        v16bf b = load_bfrag(W3p, kt, 0, 1, lane);
        c = __builtin_amdgcn_wmma_f32_16x16x32_bf16(false, a[kt], false, b,
                                                    (short)0, c, false, false);
    }
    for (int r = 0; r < 8; ++r) {
        int n = n0 + half * 8 + r;
        if (n < N && nl < 3) out[(long)n * 3 + nl] = c[r];
    }
}

// ---------------- host ----------------

extern "C" void kernel_launch(void* const* d_in, const int* in_sizes, int n_in,
                              void* d_out, int out_size, void* d_ws, size_t ws_size,
                              hipStream_t stream) {
    const float* node_features = (const float*)d_in[0];
    const float* edge_features = (const float*)d_in[1];
    const int*   edge_index    = (const int*)d_in[2];
    const int N = in_sizes[0] / 12;
    const int E = in_sizes[1] / 3;
    const int* sender   = edge_index;
    const int* receiver = edge_index + E;

    const float* enc_n_W1 = (const float*)d_in[3];
    const float* enc_n_b1 = (const float*)d_in[4];
    const float* enc_n_W2 = (const float*)d_in[5];
    const float* enc_n_b2 = (const float*)d_in[6];
    const float* enc_n_W3 = (const float*)d_in[7];
    const float* enc_n_b3 = (const float*)d_in[8];
    const float* enc_n_g  = (const float*)d_in[9];
    const float* enc_n_bt = (const float*)d_in[10];
    const float* enc_e_W1 = (const float*)d_in[11];
    const float* enc_e_b1 = (const float*)d_in[12];
    const float* enc_e_W2 = (const float*)d_in[13];
    const float* enc_e_b2 = (const float*)d_in[14];
    const float* enc_e_W3 = (const float*)d_in[15];
    const float* enc_e_b3 = (const float*)d_in[16];
    const float* enc_e_g  = (const float*)d_in[17];
    const float* enc_e_bt = (const float*)d_in[18];
    const float* pe_W1 = (const float*)d_in[19];
    const float* pe_b1 = (const float*)d_in[20];
    const float* pe_W2 = (const float*)d_in[21];
    const float* pe_b2 = (const float*)d_in[22];
    const float* pe_W3 = (const float*)d_in[23];
    const float* pe_b3 = (const float*)d_in[24];
    const float* pe_g  = (const float*)d_in[25];
    const float* pe_bt = (const float*)d_in[26];
    const float* pn_W1 = (const float*)d_in[27];
    const float* pn_b1 = (const float*)d_in[28];
    const float* pn_W2 = (const float*)d_in[29];
    const float* pn_b2 = (const float*)d_in[30];
    const float* pn_W3 = (const float*)d_in[31];
    const float* pn_b3 = (const float*)d_in[32];
    const float* pn_g  = (const float*)d_in[33];
    const float* pn_bt = (const float*)d_in[34];
    const float* dec_W1 = (const float*)d_in[35];
    const float* dec_b1 = (const float*)d_in[36];
    const float* dec_W2 = (const float*)d_in[37];
    const float* dec_b2 = (const float*)d_in[38];
    const float* dec_W3 = (const float*)d_in[39];
    const float* dec_b3 = (const float*)d_in[40];

    char* ws = (char*)d_ws;
    size_t off = 0;
    auto alloc = [&](size_t bytes) -> void* {
        void* p = ws + off;
        off = (off + bytes + 255) & ~(size_t)255;
        return p;
    };
    float* hn  = (float*)alloc((size_t)N * 128 * 4);
    float* he  = (float*)alloc((size_t)E * 128 * 4);
    float* agg = (float*)alloc((size_t)N * 128 * 4);
    auto allocW = [&](int layers, int KT, int NT) -> u32* {
        return (u32*)alloc((size_t)layers * KT * NT * 256 * 4);
    };
    u32* encN_W1p = allocW(1, 1, 8);
    u32* encN_W2p = allocW(1, 4, 8);
    u32* encN_W3p = allocW(1, 4, 8);
    u32* encE_W1p = allocW(1, 1, 8);
    u32* encE_W2p = allocW(1, 4, 8);
    u32* encE_W3p = allocW(1, 4, 8);
    u32* pe_W1p = allocW(15, 12, 8);
    u32* pe_W2p = allocW(15, 4, 8);
    u32* pe_W3p = allocW(15, 4, 8);
    u32* pn_W1p = allocW(15, 8, 8);
    u32* pn_W2p = allocW(15, 4, 8);
    u32* pn_W3p = allocW(15, 4, 8);
    u32* dec_W1p = allocW(1, 4, 8);
    u32* dec_W2p = allocW(1, 4, 8);
    u32* dec_W3p = allocW(1, 4, 1);

    auto pack = [&](const float* src, u32* dst, int layers, int Ksrc, int Kpad,
                    int Nsrc, int Npad) {
        long total = (long)layers * (Kpad / 32) * (Npad / 16) * 256;
        int blocks = (int)((total + 255) / 256);
        if (blocks > 65535) blocks = 65535;
        pack_weights_kernel<<<blocks, 256, 0, stream>>>(src, dst, layers, Ksrc, Kpad,
                                                        Nsrc, Npad);
    };
    pack(enc_n_W1, encN_W1p, 1, 12, 32, 128, 128);
    pack(enc_n_W2, encN_W2p, 1, 128, 128, 128, 128);
    pack(enc_n_W3, encN_W3p, 1, 128, 128, 128, 128);
    pack(enc_e_W1, encE_W1p, 1, 3, 32, 128, 128);
    pack(enc_e_W2, encE_W2p, 1, 128, 128, 128, 128);
    pack(enc_e_W3, encE_W3p, 1, 128, 128, 128, 128);
    pack(pe_W1, pe_W1p, 15, 384, 384, 128, 128);
    pack(pe_W2, pe_W2p, 15, 128, 128, 128, 128);
    pack(pe_W3, pe_W3p, 15, 128, 128, 128, 128);
    pack(pn_W1, pn_W1p, 15, 256, 256, 128, 128);
    pack(pn_W2, pn_W2p, 15, 128, 128, 128, 128);
    pack(pn_W3, pn_W3p, 15, 128, 128, 128, 128);
    pack(dec_W1, dec_W1p, 1, 128, 128, 128, 128);
    pack(dec_W2, dec_W2p, 1, 128, 128, 128, 128);
    pack(dec_W3, dec_W3p, 1, 128, 128, 3, 16);

    int nodeTiles = (N + 15) / 16, nodeBlocks = (nodeTiles + 1) / 2;
    int edgeTiles = (E + 15) / 16, edgeBlocks = (edgeTiles + 1) / 2;

    encoder_kernel<<<nodeBlocks, 64, 0, stream>>>(node_features, 12,
        encN_W1p, enc_n_b1, encN_W2p, enc_n_b2, encN_W3p, enc_n_b3,
        enc_n_g, enc_n_bt, hn, N);
    encoder_kernel<<<edgeBlocks, 64, 0, stream>>>(edge_features, 3,
        encE_W1p, enc_e_b1, encE_W2p, enc_e_b2, encE_W3p, enc_e_b3,
        enc_e_g, enc_e_bt, he, E);

    long aggN = (long)N * 128;
    int zBlocks = (int)((aggN + 255) / 256);
    for (int l = 0; l < 15; ++l) {
        zero_kernel<<<zBlocks, 256, 0, stream>>>(agg, aggN);
        edge_layer_kernel<<<edgeBlocks, 64, 0, stream>>>(hn, he, agg, sender, receiver,
            pe_W1p + (size_t)l * 12 * 8 * 256, pe_b1 + (size_t)l * 128,
            pe_W2p + (size_t)l * 4 * 8 * 256,  pe_b2 + (size_t)l * 128,
            pe_W3p + (size_t)l * 4 * 8 * 256,  pe_b3 + (size_t)l * 128,
            pe_g + (size_t)l * 128, pe_bt + (size_t)l * 128, E);
        node_layer_kernel<<<nodeBlocks, 64, 0, stream>>>(hn, agg,
            pn_W1p + (size_t)l * 8 * 8 * 256, pn_b1 + (size_t)l * 128,
            pn_W2p + (size_t)l * 4 * 8 * 256, pn_b2 + (size_t)l * 128,
            pn_W3p + (size_t)l * 4 * 8 * 256, pn_b3 + (size_t)l * 128,
            pn_g + (size_t)l * 128, pn_bt + (size_t)l * 128, N);
    }
    decoder_kernel<<<nodeBlocks, 64, 0, stream>>>(hn, dec_W1p, dec_b1, dec_W2p, dec_b2,
                                                  dec_W3p, dec_b3, (float*)d_out, N);
}
